// CLAM_SB_44616120271634
// MI455X (gfx1250) — matrix operand order repaired and checked by reading, not compiled, not run on hardware
//
#include <hip/hip_runtime.h>
#include <math.h>

// CLAM-SB forward for MI455X (gfx1250): bf16 WMMA GEMMs + fused epilogues.
// Heavy GEMMs -> v_wmma_f32_16x16x32_bf16 (wave32). Async global->LDS staging.

#define NROWS 50000
#define LDIM  1024
#define D1    512
#define D2    256
#define KSEL  64
#define TDIM  49
#define CCLS  2

#if __has_builtin(__builtin_amdgcn_global_load_async_to_lds_b128)
#define HAVE_ASYNC_LDS 1
#else
#define HAVE_ASYNC_LDS 0
#endif

typedef __attribute__((ext_vector_type(16))) __bf16 v16bf;
typedef __attribute__((ext_vector_type(8)))  float  v8f;
typedef int v4i_ __attribute__((vector_size(16)));
typedef __attribute__((address_space(1))) v4i_* g_v4i_ptr;
typedef __attribute__((address_space(3))) v4i_* l_v4i_ptr;

struct Frag { union { unsigned int u[8]; v16bf v; }; };

// hardware bf16 converts (v_cvt_*bf16*) instead of manual bit rounding
__device__ __forceinline__ unsigned short f2bf(float f) {
  union { __bf16 b; unsigned short u; } c;
  c.b = (__bf16)f;
  return c.u;
}
__device__ __forceinline__ unsigned int f2bf2(float lo, float hi) {
  union { __bf16 b[2]; unsigned int u; } c;
  c.b[0] = (__bf16)lo;
  c.b[1] = (__bf16)hi;
  return c.u;
}
__device__ __forceinline__ float bf2f(unsigned short h) {
  return __uint_as_float(((unsigned int)h) << 16);
}

__device__ __forceinline__ void wait_async_lds() {
#if HAVE_ASYNC_LDS
#if __has_builtin(__builtin_amdgcn_s_wait_asynccnt)
  __builtin_amdgcn_s_wait_asynccnt(0);
#else
  asm volatile("s_wait_asynccnt 0" ::: "memory");
#endif
#endif
}

// ---------------------------------------------------------------------------
// Kernel 1: h512 = relu(h @ W_fc + b_fc)   [50000,1024]x[1024,512] -> bf16
// Block 256 threads (8 waves); tile 128 rows x 64 cols; K loop 32 steps of 32.
// ---------------------------------------------------------------------------
__global__ __launch_bounds__(256) void gemm_fc_kernel(
    const float* __restrict__ h, const float* __restrict__ Wfc,
    const float* __restrict__ bfc, unsigned short* __restrict__ h512) {
  __shared__ unsigned short As[128][34];   // rows x K halves (pad 2)
  __shared__ unsigned short Bt[64][34];    // cols x K halves (transposed stage)

  const int tid  = threadIdx.x;
  const int wave = tid >> 5;
  const int lane = tid & 31;
  const int rowBase = blockIdx.x * 128;
  const int colBase = blockIdx.y * 64;
  const int m  = lane & 15;
  const int hi = lane >> 4;

  v8f zeroV = {};
  v8f acc[4] = {zeroV, zeroV, zeroV, zeroV};

  for (int kb = 0; kb < LDIM / 32; ++kb) {
    const int k0 = kb * 32;
    __syncthreads();
    // stage A: 128x32 fp32 -> bf16 (each thread: 4 float4 loads)
    {
      const int kq = (tid & 7) * 4;
      #pragma unroll
      for (int it = 0; it < 4; ++it) {
        const int lrow = it * 32 + (tid >> 3);
        const int grow = rowBase + lrow;
        float4 v = make_float4(0.f, 0.f, 0.f, 0.f);
        if (grow < NROWS)
          v = *reinterpret_cast<const float4*>(h + (size_t)grow * LDIM + k0 + kq);
        *reinterpret_cast<unsigned int*>(&As[lrow][kq])     = f2bf2(v.x, v.y);
        *reinterpret_cast<unsigned int*>(&As[lrow][kq + 2]) = f2bf2(v.z, v.w);
        if (grow < NROWS && k0 + 32 < LDIM)                 // gfx1250 global_prefetch_b8
          __builtin_prefetch(h + (size_t)grow * LDIM + k0 + 32 + kq, 0, 1);
      }
    }
    // stage B transposed: Wfc[k0..k0+31][colBase..+63] -> Bt[col][k]
    {
      const int cq = (tid & 15) * 4;
      #pragma unroll
      for (int it = 0; it < 2; ++it) {
        const int kk = it * 16 + (tid >> 4);
        float4 v = *reinterpret_cast<const float4*>(Wfc + (size_t)(k0 + kk) * D1 + colBase + cq);
        Bt[cq + 0][kk] = f2bf(v.x);
        Bt[cq + 1][kk] = f2bf(v.y);
        Bt[cq + 2][kk] = f2bf(v.z);
        Bt[cq + 3][kk] = f2bf(v.w);
      }
    }
    __syncthreads();

    // load A fragment and ALL B fragments first, then WMMAs back-to-back
    Frag af;
    {
      const int r = wave * 16 + m;
      const int kbase = hi * 16;
      #pragma unroll
      for (int j = 0; j < 8; ++j) {
        const int koff = kbase + 2 * j;
        af.u[j] = *reinterpret_cast<const unsigned int*>(&As[r][koff]);
      }
    }
    Frag bfs[4];
    #pragma unroll
    for (int ct = 0; ct < 4; ++ct) {
      const int c   = ct * 16 + m;          // B: lane n = column
      const int kbB = hi * 16;
      #pragma unroll
      for (int j = 0; j < 8; ++j)
        bfs[ct].u[j] = *reinterpret_cast<const unsigned int*>(&Bt[c][kbB + 2 * j]);
    }
    #pragma unroll
    for (int ct = 0; ct < 4; ++ct)
      acc[ct] = __builtin_amdgcn_wmma_f32_16x16x32_bf16(false, af.v, false, bfs[ct].v,
                                                        (short)0, acc[ct], false, false);
  }

  // epilogue: bias + ReLU + bf16 store. C layout: VGPR v -> row v + 8*hi.
  #pragma unroll
  for (int ct = 0; ct < 4; ++ct) {
    const int col = colBase + ct * 16 + m;
    const float bias = bfc[col];
    #pragma unroll
    for (int v = 0; v < 8; ++v) {
      const int row = rowBase + wave * 16 + v + 8 * hi;
      if (row < NROWS) {
        float x = fmaxf(acc[ct][v] + bias, 0.0f);
        h512[(size_t)row * D1 + col] = f2bf(x);
      }
    }
  }
}

// ---------------------------------------------------------------------------
// Kernel 2: fused gated attention:
//   a = tanh(h512@Wa+ba), b = sigmoid(h512@Wb+bb), s = (a*b)@Wc + bc
// Block 256 threads; tile 32 rows x 256 cols (2 row-groups x 4 col-groups).
// A tile staged with async global->LDS copies when available (bf16 passthrough).
// ---------------------------------------------------------------------------
__global__ __launch_bounds__(256) void gemm_attn_kernel(
    const unsigned short* __restrict__ h512,
    const float* __restrict__ Wa, const float* __restrict__ ba,
    const float* __restrict__ Wb, const float* __restrict__ bb,
    const float* __restrict__ Wc, const float* __restrict__ bc,
    float* __restrict__ scoresA, float* __restrict__ scoresB,
    float* __restrict__ Araw) {
  __shared__ unsigned short As[32][40];     // 80B row stride: 16B-aligned rows
  __shared__ unsigned short Wat[256][34];
  __shared__ unsigned short Wbt[256][34];
  __shared__ float red[4][32];

  const int tid  = threadIdx.x;
  const int wave = tid >> 5;
  const int lane = tid & 31;
  const int rg = wave >> 2;                 // row group 0..1
  const int cg = wave & 3;                  // col group 0..3
  const int rowBase = blockIdx.x * 32;
  const int m  = lane & 15;
  const int hi = lane >> 4;

  // pre-zero A tile once: async path skips OOB rows, they must read as zero
  for (int i = tid; i < 32 * 40 / 2; i += 256)
    reinterpret_cast<unsigned int*>(&As[0][0])[i] = 0u;

  v8f zeroV = {};
  v8f accA[4] = {zeroV, zeroV, zeroV, zeroV};
  v8f accB[4] = {zeroV, zeroV, zeroV, zeroV};

  for (int kb = 0; kb < D1 / 32; ++kb) {
    const int k0 = kb * 32;
    __syncthreads();
#if HAVE_ASYNC_LDS
    // stage A asynchronously: 32 rows x 64B, one b128 per thread (tid<128)
    if (tid < 128) {
      const int lrow = tid >> 2;
      const int seg  = tid & 3;
      const int grow = rowBase + lrow;
      if (grow < NROWS) {
        unsigned short* gp =
            const_cast<unsigned short*>(h512 + (size_t)grow * D1 + k0 + seg * 8);
        __builtin_amdgcn_global_load_async_to_lds_b128(
            (g_v4i_ptr)(void*)gp,
            (l_v4i_ptr)(void*)&As[lrow][seg * 8],
            0, 0);
      }
    }
#else
    // stage A: 32 rows x 32 bf16 halves via VGPR copy
    #pragma unroll
    for (int it = 0; it < 2; ++it) {
      const int i    = it * 256 + tid;      // 512 dwords total
      const int lrow = i >> 4;
      const int kd   = i & 15;
      const int grow = rowBase + lrow;
      unsigned int v = 0;
      if (grow < NROWS)
        v = *reinterpret_cast<const unsigned int*>(h512 + (size_t)grow * D1 + k0 + kd * 2);
      *reinterpret_cast<unsigned int*>(&As[lrow][kd * 2]) = v;
    }
#endif
    // stage Wa/Wb transposed: 32 x 256 fp32 each -> [col][k] bf16
    {
      const int cq = (tid & 63) * 4;
      #pragma unroll
      for (int it = 0; it < 8; ++it) {
        const int kk = it * 4 + (tid >> 6);
        float4 va = *reinterpret_cast<const float4*>(Wa + (size_t)(k0 + kk) * D2 + cq);
        float4 vb = *reinterpret_cast<const float4*>(Wb + (size_t)(k0 + kk) * D2 + cq);
        Wat[cq + 0][kk] = f2bf(va.x); Wat[cq + 1][kk] = f2bf(va.y);
        Wat[cq + 2][kk] = f2bf(va.z); Wat[cq + 3][kk] = f2bf(va.w);
        Wbt[cq + 0][kk] = f2bf(vb.x); Wbt[cq + 1][kk] = f2bf(vb.y);
        Wbt[cq + 2][kk] = f2bf(vb.z); Wbt[cq + 3][kk] = f2bf(vb.w);
      }
    }
    wait_async_lds();                       // drain this wave's async copies
    __syncthreads();

    Frag af;
    {
      const int r = rg * 16 + m;
      const int kbase = hi * 16;
      #pragma unroll
      for (int j = 0; j < 8; ++j) {
        const int koff = kbase + 2 * j;
        af.u[j] = *reinterpret_cast<const unsigned int*>(&As[r][koff]);
      }
    }
    Frag bfa[4], bfb[4];
    #pragma unroll
    for (int ct = 0; ct < 4; ++ct) {
      const int c   = cg * 64 + ct * 16 + m;
      const int kbB = hi * 16;
      #pragma unroll
      for (int j = 0; j < 8; ++j) {
        bfa[ct].u[j] = *reinterpret_cast<const unsigned int*>(&Wat[c][kbB + 2 * j]);
        bfb[ct].u[j] = *reinterpret_cast<const unsigned int*>(&Wbt[c][kbB + 2 * j]);
      }
    }
    #pragma unroll
    for (int ct = 0; ct < 4; ++ct) {
      accA[ct] = __builtin_amdgcn_wmma_f32_16x16x32_bf16(false, af.v, false, bfa[ct].v,
                                                         (short)0, accA[ct], false, false);
      accB[ct] = __builtin_amdgcn_wmma_f32_16x16x32_bf16(false, af.v, false, bfb[ct].v,
                                                         (short)0, accB[ct], false, false);
    }
  }

  // fused epilogue: tanh * sigmoid * Wc, reduce across 256 cols
  #pragma unroll
  for (int v = 0; v < 8; ++v) {
    float s = 0.f;
    #pragma unroll
    for (int ct = 0; ct < 4; ++ct) {
      const int col = cg * 64 + ct * 16 + m;
      float av = tanhf(accA[ct][v] + ba[col]);
      float bv = accB[ct][v] + bb[col];
      bv = 1.0f / (1.0f + expf(-bv));
      s += av * bv * Wc[col];
    }
    s += __shfl_xor(s, 1, 32);              // reduce across 16-lane half-groups
    s += __shfl_xor(s, 2, 32);
    s += __shfl_xor(s, 4, 32);
    s += __shfl_xor(s, 8, 32);
    if (m == 0) red[cg][rg * 16 + hi * 8 + v] = s;
  }
  __syncthreads();
  if (tid < 32) {
    const int grow = rowBase + tid;
    if (grow < NROWS) {
      float s = red[0][tid] + red[1][tid] + red[2][tid] + red[3][tid] + bc[0];
      scoresA[grow] = s;
      scoresB[grow] = s;
      Araw[grow]    = s;
    }
  }
}

// ---------------------------------------------------------------------------
// Softmax statistics + pooling + top-k + instance loss + head
// ---------------------------------------------------------------------------
__global__ void zero_kernel(float* __restrict__ p, int n) {
  int i = blockIdx.x * blockDim.x + threadIdx.x;
  if (i < n) p[i] = 0.f;
}

__global__ __launch_bounds__(256) void partial_max_kernel(
    const float* __restrict__ s, float* __restrict__ pmax) {
  __shared__ float sm[256];
  const int tid = threadIdx.x;
  float mx = -__builtin_inff();
  for (int i = blockIdx.x * 256 + tid; i < NROWS; i += 256 * 64) mx = fmaxf(mx, s[i]);
  sm[tid] = mx; __syncthreads();
  for (int off = 128; off > 0; off >>= 1) {
    if (tid < off) sm[tid] = fmaxf(sm[tid], sm[tid + off]);
    __syncthreads();
  }
  if (tid == 0) pmax[blockIdx.x] = sm[0];
}

__global__ __launch_bounds__(256) void partial_sum_kernel(
    const float* __restrict__ s, const float* __restrict__ pmax,
    float* __restrict__ psum) {
  __shared__ float sm[256];
  const int tid = threadIdx.x;
  float g = -__builtin_inff();
  for (int i = 0; i < 64; ++i) g = fmaxf(g, pmax[i]);
  float acc = 0.f;
  for (int i = blockIdx.x * 256 + tid; i < NROWS; i += 256 * 64) acc += expf(s[i] - g);
  sm[tid] = acc; __syncthreads();
  for (int off = 128; off > 0; off >>= 1) {
    if (tid < off) sm[tid] += sm[tid + off];
    __syncthreads();
  }
  if (tid == 0) psum[blockIdx.x] = sm[0];
}

__global__ void finalize_stats_kernel(const float* __restrict__ pmax,
                                      const float* __restrict__ psum,
                                      float* __restrict__ stats) {
  if (threadIdx.x == 0) {
    float g = -__builtin_inff(), sum = 0.f;
    for (int i = 0; i < 64; ++i) { g = fmaxf(g, pmax[i]); sum += psum[i]; }
    stats[0] = g;
    stats[1] = 1.0f / sum;
  }
}

// M[d] = sum_i softmax(s)_i * h512[i,d]
__global__ __launch_bounds__(256) void pool_kernel(
    const float* __restrict__ s, const float* __restrict__ stats,
    const unsigned short* __restrict__ h512, float* __restrict__ M) {
  const int tid = threadIdx.x;
  const float g = stats[0], inv = stats[1];
  const int r0 = blockIdx.x * 256;
  const int r1 = min(r0 + 256, NROWS);
  float a0 = 0.f, a1 = 0.f;
  for (int r = r0; r < r1; ++r) {
    const float p = expf(s[r] - g) * inv;
    const unsigned short* row = h512 + (size_t)r * D1;
    a0 += p * bf2f(row[tid]);
    a1 += p * bf2f(row[tid + 256]);
  }
  atomicAdd(&M[tid], a0);
  atomicAdd(&M[tid + 256], a1);
}

// exact top-64 / bottom-64 by repeated argmax/argmin (tie -> lower index)
__global__ __launch_bounds__(256) void topk_kernel(
    float* __restrict__ sa, float* __restrict__ sb,
    int* __restrict__ topids, int* __restrict__ botids) {
  __shared__ float sv[256];
  __shared__ int   si[256];
  const int tid = threadIdx.x;
  const float INF = __builtin_inff();
  for (int pass = 0; pass < KSEL; ++pass) {
    float best = -INF; int bi = 0x7FFFFFFF;
    for (int i = tid; i < NROWS; i += 256) {
      float v = sa[i];
      if (v > best || (v == best && i < bi)) { best = v; bi = i; }
    }
    sv[tid] = best; si[tid] = bi; __syncthreads();
    for (int off = 128; off > 0; off >>= 1) {
      if (tid < off) {
        float ov = sv[tid + off]; int oi = si[tid + off];
        if (ov > sv[tid] || (ov == sv[tid] && oi < si[tid])) { sv[tid] = ov; si[tid] = oi; }
      }
      __syncthreads();
    }
    if (tid == 0) { topids[pass] = si[0]; sa[si[0]] = -INF; }
    __syncthreads();
  }
  for (int pass = 0; pass < KSEL; ++pass) {
    float best = INF; int bi = 0x7FFFFFFF;
    for (int i = tid; i < NROWS; i += 256) {
      float v = sb[i];
      if (v < best || (v == best && i < bi)) { best = v; bi = i; }
    }
    sv[tid] = best; si[tid] = bi; __syncthreads();
    for (int off = 128; off > 0; off >>= 1) {
      if (tid < off) {
        float ov = sv[tid + off]; int oi = si[tid + off];
        if (ov < sv[tid] || (ov == sv[tid] && oi < si[tid])) { sv[tid] = ov; si[tid] = oi; }
      }
      __syncthreads();
    }
    if (tid == 0) { botids[pass] = si[0]; sb[si[0]] = INF; }
    __syncthreads();
  }
}

// instance CE loss for the in-class classifier only (one-hot collapses vmap)
__global__ __launch_bounds__(128) void inst_loss_kernel(
    const unsigned short* __restrict__ h512, const int* __restrict__ topids,
    const int* __restrict__ botids, const float* __restrict__ Winst,
    const float* __restrict__ binst, const int* __restrict__ label,
    float* __restrict__ out_loss) {
  __shared__ float red[128];
  const int tid = threadIdx.x;
  const int lab = label[0];
  const float* W = Winst + (size_t)lab * D1 * 2;
  const float* b = binst + (size_t)lab * 2;
  const int id     = (tid < KSEL) ? topids[tid] : botids[tid - KSEL];
  const int target = (tid < KSEL) ? 1 : 0;
  float l0 = b[0], l1 = b[1];
  const unsigned short* row = h512 + (size_t)id * D1;
  for (int d = 0; d < D1; ++d) {
    float x = bf2f(row[d]);
    l0 += x * W[d * 2 + 0];
    l1 += x * W[d * 2 + 1];
  }
  float mx  = fmaxf(l0, l1);
  float lse = mx + logf(expf(l0 - mx) + expf(l1 - mx));
  float logp = (target ? l1 : l0) - lse;
  red[tid] = -logp;
  __syncthreads();
  for (int off = 64; off > 0; off >>= 1) {
    if (tid < off) red[tid] += red[tid + off];
    __syncthreads();
  }
  if (tid == 0) out_loss[0] = red[0] / (2.0f * KSEL);
}

// 1-head "MHA" gating + bag classifier
__global__ __launch_bounds__(64) void head_kernel(
    const float* __restrict__ M, const float* __restrict__ tab,
    const float* __restrict__ Wimg, const float* __restrict__ bimg,
    const float* __restrict__ Wtab, const float* __restrict__ btab,
    const float* __restrict__ Wcls, const float* __restrict__ bcls,
    float* __restrict__ out) {
  __shared__ float r0[64], r1[64];
  __shared__ float tl[TDIM], tsm[TDIM];
  __shared__ float sh_img;
  const int tid = threadIdx.x;

  float acc = 0.f;
  for (int d = tid; d < D1; d += 64) acc += M[d] * Wimg[d];
  r0[tid] = acc; __syncthreads();
  for (int off = 32; off > 0; off >>= 1) {
    if (tid < off) r0[tid] += r0[tid + off];
    __syncthreads();
  }
  if (tid == 0) sh_img = 1.0f / (1.0f + expf(-(r0[0] + bimg[0])));

  if (tid < TDIM) {
    float a = btab[tid];
    for (int i = 0; i < TDIM; ++i) a += tab[i] * Wtab[i * TDIM + tid];
    tl[tid] = a;
  }
  __syncthreads();
  if (tid == 0) {
    float mx = -__builtin_inff();
    for (int j = 0; j < TDIM; ++j) mx = fmaxf(mx, tl[j]);
    float sum = 0.f;
    for (int j = 0; j < TDIM; ++j) { tsm[j] = expf(tl[j] - mx); sum += tsm[j]; }
    float inv = 1.0f / sum;
    for (int j = 0; j < TDIM; ++j) tsm[j] *= inv;
  }
  __syncthreads();
  const float img = sh_img;

  float a0 = 0.f, a1 = 0.f;
  for (int i = tid; i < D1 + TDIM; i += 64) {
    float c = (i < D1) ? img * M[i] : tsm[i - D1] * tab[i - D1];
    a0 += c * Wcls[i * 2 + 0];
    a1 += c * Wcls[i * 2 + 1];
  }
  r0[tid] = a0; r1[tid] = a1; __syncthreads();
  for (int off = 32; off > 0; off >>= 1) {
    if (tid < off) { r0[tid] += r0[tid + off]; r1[tid] += r1[tid + off]; }
    __syncthreads();
  }
  if (tid == 0) {
    float l0 = r0[0] + bcls[0];
    float l1 = r1[0] + bcls[1];
    out[0] = l0; out[1] = l1;
    float mx = fmaxf(l0, l1);
    float e0 = expf(l0 - mx), e1 = expf(l1 - mx);
    float inv = 1.0f / (e0 + e1);
    out[2] = e0 * inv;
    out[3] = e1 * inv;
    out[4] = (l1 > l0) ? 1.0f : 0.0f;     // argmax, first-max on ties
  }
}

// ---------------------------------------------------------------------------
extern "C" void kernel_launch(void* const* d_in, const int* in_sizes, int n_in,
                              void* d_out, int out_size, void* d_ws, size_t ws_size,
                              hipStream_t stream) {
  const float* h     = (const float*)d_in[0];
  const float* tab   = (const float*)d_in[1];
  const int*   lab   = (const int*)  d_in[2];
  const float* Wfc   = (const float*)d_in[3];
  const float* bfc   = (const float*)d_in[4];
  const float* Wa    = (const float*)d_in[5];
  const float* ba    = (const float*)d_in[6];
  const float* Wb    = (const float*)d_in[7];
  const float* bb    = (const float*)d_in[8];
  const float* Wc    = (const float*)d_in[9];
  const float* bc    = (const float*)d_in[10];
  const float* Winst = (const float*)d_in[11];
  const float* binst = (const float*)d_in[12];
  const float* Wimg  = (const float*)d_in[13];
  const float* bimg  = (const float*)d_in[14];
  const float* Wtab  = (const float*)d_in[15];
  const float* btab  = (const float*)d_in[16];
  const float* Wcls  = (const float*)d_in[17];
  const float* bcls  = (const float*)d_in[18];
  float* out = (float*)d_out;

  // workspace layout
  char* ws = (char*)d_ws;
  size_t off = 0;
  unsigned short* h512 = (unsigned short*)(ws + off); off += (size_t)NROWS * D1 * 2;
  float* scoresA = (float*)(ws + off); off += (size_t)NROWS * 4;
  float* scoresB = (float*)(ws + off); off += (size_t)NROWS * 4;
  float* pmax    = (float*)(ws + off); off += 64 * 4;
  float* psum    = (float*)(ws + off); off += 64 * 4;
  float* stats   = (float*)(ws + off); off += 2 * 4;
  int*   topids  = (int*)  (ws + off); off += KSEL * 4;
  int*   botids  = (int*)  (ws + off); off += KSEL * 4;
  float* Mbuf    = (float*)(ws + off); off += D1 * 4;

  float* Araw     = out + 5;                 // [N]
  float* out_loss = out + 5 + NROWS;         // [1]

  zero_kernel<<<2, 256, 0, stream>>>(Mbuf, D1);

  dim3 g1((NROWS + 127) / 128, D1 / 64);
  gemm_fc_kernel<<<g1, 256, 0, stream>>>(h, Wfc, bfc, h512);

  gemm_attn_kernel<<<(NROWS + 31) / 32, 256, 0, stream>>>(
      h512, Wa, ba, Wb, bb, Wc, bc, scoresA, scoresB, Araw);

  partial_max_kernel<<<64, 256, 0, stream>>>(Araw, pmax);
  partial_sum_kernel<<<64, 256, 0, stream>>>(Araw, pmax, psum);
  finalize_stats_kernel<<<1, 32, 0, stream>>>(pmax, psum, stats);

  pool_kernel<<<(NROWS + 255) / 256, 256, 0, stream>>>(Araw, stats, h512, Mbuf);

  topk_kernel<<<1, 256, 0, stream>>>(scoresA, scoresB, topids, botids);

  inst_loss_kernel<<<1, 128, 0, stream>>>(h512, topids, botids, Winst, binst, lab, out_loss);

  head_kernel<<<1, 64, 0, stream>>>(Mbuf, tab, Wimg, bimg, Wtab, btab, Wcls, bcls, out);
}
